// DeformableConv2d_34205119545411
// MI455X (gfx1250) — compile-verified, matrix operand order checked by reading, not run on hardware
//
#include <hip/hip_runtime.h>
#include <math.h>

// Problem constants (from the reference setup)
#define BATCH 8
#define CIN   64
#define COUT  64
#define HH    128
#define WW    128
#define HWSZ  (HH * WW)
#define K2    9
#define RED   (K2 * CIN)   // 576 reduction length (tap-major: r = k*64 + c)
#define TILE_M 16
#define KSTEPS (RED / 32)  // 18 bf16 WMMA K-steps

typedef __attribute__((ext_vector_type(16))) __bf16 v16bf;
typedef __attribute__((ext_vector_type(8)))  float  v8f;

// Fragment-order swizzle (16-bit A/B WMMA layout, wave32):
//   lane = (row/col & 15) + half*16,  half = (kl>>3)&1
//   e    = (kl & 7) + ((kl>>4)<<3)        (kl = local K index 0..31)
// Forward map used by consumers: kl(e) = kb + (e<8 ? e : 8+e), kb = lane<16?0:8.
// Quad e0 in {0,4,8,12} -> kl0 = kb + (e0<8 ? e0 : 8+e0), kl consecutive in quad.

// ---------------------------------------------------------------------------
// x transpose NCHW -> NHWC so one bilinear corner x 4 consecutive channels is
// a single float4 (and a corner's 64 channels are 256 contiguous bytes).
// LDS tile with 129-float stride: both global directions coalesced, LDS
// conflict-free. One block per (h, b) row.
// ---------------------------------------------------------------------------
__global__ __launch_bounds__(256)
void transpose_x(const float* __restrict__ x, float* __restrict__ xt) {
    __shared__ float tile[CIN][WW + 1];
    const int h = blockIdx.x;
    const int b = blockIdx.y;
    const int t = threadIdx.x;
    const float* src = x + ((size_t)b * CIN * HH + h) * WW;      // + c*HWSZ + w
    float*       dst = xt + ((size_t)b * HH + h) * WW * CIN;     // + w*CIN + c
    for (int idx = t; idx < CIN * WW; idx += 256) {
        int c = idx >> 7, w = idx & (WW - 1);
        tile[c][w] = src[(size_t)c * HWSZ + w];                  // coalesced in w
    }
    __syncthreads();
    for (int idx = t; idx < CIN * WW; idx += 256) {
        int w = idx >> 6, c = idx & (CIN - 1);
        dst[(size_t)w * CIN + c] = tile[c][w];                   // coalesced in c
    }
}

// ---------------------------------------------------------------------------
// Weight repack: f32 OIHW -> bf16 panels pre-swizzled into fragment order.
// wbAs: [KSTEPS][2 wv][32 lane][16 e]  (n<18: w_off, 18<=n<27: w_mod, pad 0)
// wbBs: [KSTEPS][4 wv][32 lane][16 e]  (main conv weights)
// ---------------------------------------------------------------------------
__global__ void prep_weights(const float* __restrict__ w_off,
                             const float* __restrict__ w_mod,
                             const float* __restrict__ w,
                             __bf16* __restrict__ wbAs,
                             __bf16* __restrict__ wbBs) {
    int t = blockIdx.x * 256 + threadIdx.x;
    const int nA = KSTEPS * 2 * 32 * 16;   // 18432
    const int nB = KSTEPS * 4 * 32 * 16;   // 36864
    if (t < nA) {
        int e = t & 15, lane = (t >> 4) & 31, wv = (t >> 9) & 1, kk = t >> 10;
        int kb = (lane < 16) ? 0 : 8;
        int kl = kb + (e < 8 ? e : 8 + e);
        int k = kk >> 1;
        int c = (kk & 1) * 32 + kl;
        int n = wv * 16 + (lane & 15);
        float v = 0.f;
        if (n < 18)      v = w_off[(n * CIN + c) * K2 + k];
        else if (n < 27) v = w_mod[((n - 18) * CIN + c) * K2 + k];
        wbAs[t] = (__bf16)v;
    } else if (t < nA + nB) {
        int t2 = t - nA;
        int e = t2 & 15, lane = (t2 >> 4) & 31, wv = (t2 >> 9) & 3, kk = t2 >> 11;
        int kb = (lane < 16) ? 0 : 8;
        int kl = kb + (e < 8 ? e : 8 + e);
        int k = kk >> 1;
        int c = (kk & 1) * 32 + kl;
        int o = wv * 16 + (lane & 15);
        wbBs[t2] = (__bf16)w[(o * CIN + c) * K2 + k];
    }
}

// ---------------------------------------------------------------------------
// Kernel A: fused offset (18ch) + mask (9ch, 2*sigmoid) 3x3 conv via WMMA.
// One workgroup (2 waves) = 16 pixels of one row. All 18 fragment-order
// A-tiles built once from NHWC x (im2col quad = one float4), then a
// sync-free WMMA loop with direct-from-global B fragments.
// Output: planar f32 offmask[B][27][H][W] in workspace.
// ---------------------------------------------------------------------------
__global__ __launch_bounds__(64)
void offset_mask_conv(const float* __restrict__ xt,
                      const float* __restrict__ b_off,
                      const float* __restrict__ b_mod,
                      const __bf16* __restrict__ wbAs,
                      float* __restrict__ offmask) {
    __shared__ __bf16 aswz[KSTEPS][32][16];   // 18 KB, fragment order

    const int w0   = blockIdx.x * TILE_M;
    const int h    = blockIdx.y;
    const int b    = blockIdx.z;
    const int t    = threadIdx.x;
    const int lane = t & 31;
    const int waveN = t >> 5;                 // N-tile: ch 0..15 / 16..31

    const float* xtb = xt + (size_t)b * HWSZ * CIN;

    // Build fragment-order A tiles: thread owns (lane_t, 8 e-values) x all kk.
    {
        const int lane_t = t >> 1;
        const int mt  = lane_t & 15;
        const int kb  = (lane_t < 16) ? 0 : 8;
        const int eh  = (t & 1) * 8;
        for (int kk = 0; kk < KSTEPS; ++kk) {
            const int k_tap = kk >> 1;
            const int ki = k_tap / 3, kj = k_tap % 3;
            const int cb = (kk & 1) * 32;
            const int gh = h - 1 + ki;
            const int gw = w0 - 1 + mt + kj;
            const bool valid = (gh >= 0) && (gh < HH) && (gw >= 0) && (gw < WW);
            const float* xs = xtb + ((size_t)gh * WW + gw) * CIN + cb;
#pragma unroll
            for (int q = 0; q < 2; ++q) {
                const int e0  = eh + q * 4;
                const int kl0 = kb + (e0 < 8 ? e0 : 8 + e0);
                float4 v4 = make_float4(0.f, 0.f, 0.f, 0.f);
                if (valid) v4 = *(const float4*)(xs + kl0);      // 4 channels
                union { __bf16 hv[4]; uint2 u2; } pkt;
                pkt.hv[0] = (__bf16)v4.x;
                pkt.hv[1] = (__bf16)v4.y;
                pkt.hv[2] = (__bf16)v4.z;
                pkt.hv[3] = (__bf16)v4.w;
                *(uint2*)&aswz[kk][lane_t][e0] = pkt.u2;         // ds_store_b64
            }
        }
    }
    __syncthreads();

    v8f acc = {};
    for (int kk = 0; kk < KSTEPS; ++kk) {
        v16bf af = *(const v16bf*)&aswz[kk][lane][0];            // 2x ds_load_b128
        v16bf bf = *(const v16bf*)(wbAs + (size_t)((kk * 2 + waveN) * 32 + lane) * 16);
        if (kk + 1 < KSTEPS)
            __builtin_prefetch(wbAs + (size_t)(((kk + 1) * 2 + waveN) * 32 + lane) * 16, 0, 1);
        acc = __builtin_amdgcn_wmma_f32_16x16x32_bf16(
                  false, af, false, bf, (short)0, acc, false, false);
    }

    // Epilogue: bias, 2*sigmoid on mask channels, planar f32 store
    const int n = waveN * 16 + (lane & 15);
    float bias = (n < 18) ? b_off[n] : (n < 27 ? b_mod[n - 18] : 0.f);
    float* om = offmask + (size_t)b * 27 * HWSZ + (size_t)h * WW;
#pragma unroll
    for (int r = 0; r < 8; ++r) {
        int mp = r + ((lane < 16) ? 0 : 8);   // pixel index per C/D layout
        float v = acc[r] + bias;
        if (n < 18) {
            om[(size_t)n * HWSZ + w0 + mp] = v;
        } else if (n < 27) {
            om[(size_t)n * HWSZ + w0 + mp] = 2.f / (1.f + __expf(-v));
        }
    }
}

// ---------------------------------------------------------------------------
// Kernel B: fused bilinear sampling + [16 x 576] @ [576 x 64] WMMA GEMM.
// 4 waves; wave n owns output channels 16n..16n+15. Bilinear corner pixel
// indices and mask-scaled weights (zeroed when invalid, matching the
// reference) precomputed per (pixel,tap) as int4/float4. A-tile build reads
// NHWC x: one corner x 4 channels = one float4. Then a sync-free WMMA loop.
// ---------------------------------------------------------------------------
__global__ __launch_bounds__(128)
void deform_gemm(const float* __restrict__ xt,
                 const float* __restrict__ bias,
                 const __bf16* __restrict__ wbBs,
                 const float* __restrict__ offmask,
                 float* __restrict__ out) {
    __shared__ int4   s_off[TILE_M * K2];
    __shared__ float4 s_w  [TILE_M * K2];
    __shared__ __bf16 aswz[KSTEPS][32][16];   // 18 KB, fragment order

    const int w0   = blockIdx.x * TILE_M;
    const int h    = blockIdx.y;
    const int b    = blockIdx.z;
    const int t    = threadIdx.x;
    const int lane = t & 31;
    const int waveN = t >> 5;                 // 0..3 -> output channel tile

    const float* xtb = xt + (size_t)b * HWSZ * CIN;
    const float* om  = offmask + (size_t)b * 27 * HWSZ + (size_t)h * WW;

    // Per-(pixel,tap) bilinear parameters (144 entries); offsets are pixel
    // indices into the NHWC plane (byte addr = idx*256 + c*4).
    for (int pk = t; pk < TILE_M * K2; pk += 128) {
        int p = pk / K2, k = pk % K2;
        int ki = k / 3, kj = k % 3;
        int wpix = w0 + p;
        float dy = om[(size_t)(2 * k) * HWSZ + wpix];
        float dx = om[(size_t)(2 * k + 1) * HWSZ + wpix];
        float mv = om[(size_t)(18 + k) * HWSZ + wpix];
        float py = dy + (float)(h - 1 + ki);
        float px = dx + (float)(wpix - 1 + kj);
        float y0f = floorf(py), x0f = floorf(px);
        float wy = py - y0f, wx = px - x0f;
        int y0 = (int)y0f, x0 = (int)x0f;
        int   oc[4];
        float wc[4];
#pragma unroll
        for (int c4 = 0; c4 < 4; ++c4) {
            int yi = y0 + (c4 >> 1), xi = x0 + (c4 & 1);
            bool valid = (yi >= 0) && (yi < HH) && (xi >= 0) && (xi < WW);
            int yc = min(max(yi, 0), HH - 1);
            int xc = min(max(xi, 0), WW - 1);
            float wgt = ((c4 >> 1) ? wy : 1.f - wy) * ((c4 & 1) ? wx : 1.f - wx);
            oc[c4] = yc * WW + xc;
            wc[c4] = valid ? wgt * mv : 0.f;   // mask fused into weight
        }
        s_off[pk] = make_int4(oc[0], oc[1], oc[2], oc[3]);
        s_w[pk]   = make_float4(wc[0], wc[1], wc[2], wc[3]);
    }
    __syncthreads();

    // Build fragment-order A tiles: thread owns (lane_t, e0..e0+3) x all taps.
    // Per tap-half: 4 corner float4 gathers -> weighted sum -> packed b64 store.
    {
        const int lane_t = t >> 2;
        const int mt  = lane_t & 15;
        const int kb  = (lane_t < 16) ? 0 : 8;
        const int e0  = (t & 3) * 4;
        const int kl0 = kb + (e0 < 8 ? e0 : 8 + e0);
        for (int k_tap = 0; k_tap < K2; ++k_tap) {
            int4   o4 = s_off[mt * K2 + k_tap];   // 1x ds_load_b128
            float4 w4 = s_w[mt * K2 + k_tap];     // 1x ds_load_b128
#pragma unroll
            for (int half = 0; half < 2; ++half) {
                const int kk = k_tap * 2 + half;
                const float* base = xtb + half * 32 + kl0;
                float4 g0 = *(const float4*)(base + (size_t)o4.x * CIN);
                float4 g1 = *(const float4*)(base + (size_t)o4.y * CIN);
                float4 g2 = *(const float4*)(base + (size_t)o4.z * CIN);
                float4 g3 = *(const float4*)(base + (size_t)o4.w * CIN);
                union { __bf16 hv[4]; uint2 u2; } pkt;
                pkt.hv[0] = (__bf16)(w4.x * g0.x + w4.y * g1.x + w4.z * g2.x + w4.w * g3.x);
                pkt.hv[1] = (__bf16)(w4.x * g0.y + w4.y * g1.y + w4.z * g2.y + w4.w * g3.y);
                pkt.hv[2] = (__bf16)(w4.x * g0.z + w4.y * g1.z + w4.z * g2.z + w4.w * g3.z);
                pkt.hv[3] = (__bf16)(w4.x * g0.w + w4.y * g1.w + w4.z * g2.w + w4.w * g3.w);
                *(uint2*)&aswz[kk][lane_t][e0] = pkt.u2;   // ds_store_b64
            }
        }
    }
    __syncthreads();

    // Sync-free WMMA loop: 2x ds_load_b128 (A) + 32B global (B) + wmma
    v8f acc = {};
    for (int kk = 0; kk < KSTEPS; ++kk) {
        v16bf af = *(const v16bf*)&aswz[kk][lane][0];
        v16bf bf = *(const v16bf*)(wbBs + (size_t)((kk * 4 + waveN) * 32 + lane) * 16);
        if (kk + 1 < KSTEPS)
            __builtin_prefetch(wbBs + (size_t)(((kk + 1) * 4 + waveN) * 32 + lane) * 16, 0, 1);
        acc = __builtin_amdgcn_wmma_f32_16x16x32_bf16(
                  false, af, false, bf, (short)0, acc, false, false);
    }

    // Epilogue: bias + NCHW store (contiguous-pixel stores per lane)
    const int o = waveN * 16 + (lane & 15);
    float bo = bias[o];
    float* outp = out + ((size_t)(b * COUT + o) * HH + h) * WW + w0;
#pragma unroll
    for (int r = 0; r < 8; ++r) {
        int mp = r + ((lane < 16) ? 0 : 8);
        outp[mp] = acc[r] + bo;
    }
}

// ---------------------------------------------------------------------------
// Launch. Workspace layout (~47.8 MB total):
//   [0)         wbAs : 576*32 bf16 (fragment order)          36,864 B
//   [36864)     wbBs : 576*64 bf16 (fragment order)          73,728 B
//   [110592)    offmask : 8*27*128*128 f32               14,155,776 B
//   [14266368)  xt : NHWC x, 8*128*128*64 f32            33,554,432 B
// ---------------------------------------------------------------------------
extern "C" void kernel_launch(void* const* d_in, const int* in_sizes, int n_in,
                              void* d_out, int out_size, void* d_ws, size_t ws_size,
                              hipStream_t stream) {
    const float* x     = (const float*)d_in[0];
    const float* w_off = (const float*)d_in[1];
    const float* b_off = (const float*)d_in[2];
    const float* w_mod = (const float*)d_in[3];
    const float* b_mod = (const float*)d_in[4];
    const float* w     = (const float*)d_in[5];
    const float* bias  = (const float*)d_in[6];
    float* out = (float*)d_out;

    char* ws = (char*)d_ws;
    __bf16* wbAs = (__bf16*)ws;
    __bf16* wbBs = (__bf16*)(ws + 36864);
    float*  offmask = (float*)(ws + 110592);
    float*  xt      = (float*)(ws + 14266368);

    const int prepN = KSTEPS * 2 * 32 * 16 + KSTEPS * 4 * 32 * 16;
    prep_weights<<<(prepN + 255) / 256, 256, 0, stream>>>(w_off, w_mod, w, wbAs, wbBs);
    transpose_x<<<dim3(HH, BATCH), 256, 0, stream>>>(x, xt);

    dim3 grid(WW / TILE_M, HH, BATCH);
    offset_mask_conv<<<grid, 64, 0, stream>>>(xt, b_off, b_mod, wbAs, offmask);
    deform_gemm<<<grid, 128, 0, stream>>>(xt, bias, wbBs, offmask, out);
}
